// CosineSimilarityLayer_64570538328853
// MI455X (gfx1250) — matrix operand order repaired and checked by reading, not compiled, not run on hardware
//
#include <hip/hip_runtime.h>

// ---------------------------------------------------------------------------
// Cosine similarity layer: out[b,n] = <x[b,:]/|x[b,:]|, Wn[:,n]>,
// Wn rows (per input-dim d) normalized over N.
// Split-f16 (hi/lo, lo scaled by 2^12) GEMM on v_wmma_f32_16x16x32_f16,
// double-buffered LDS fed by GLOBAL_LOAD_ASYNC_TO_LDS_B128 (ASYNCcnt).
// ---------------------------------------------------------------------------

typedef __attribute__((ext_vector_type(8)))  _Float16 v8h;
typedef __attribute__((ext_vector_type(16))) _Float16 v16h;
typedef __attribute__((ext_vector_type(8)))  float    v8f;

#define B_DIM 4096
#define D_DIM 512
#define N_DIM 5994
#define N_PAD 6016   // 94 tiles of 64
#define TM    128
#define TN    64
#define LDSTR 40     // padded LDS row stride in f16 elems (80B, conflict-free)
#define KSTEPS (D_DIM / 32)
#define LO_SCALE      4096.0f
#define LO_SCALE_INV  (1.0f / 4096.0f)

static __device__ __forceinline__ v16h join8(v8h a, v8h b) {
  union { v16h v; v8h h[2]; } u;
  u.h[0] = a; u.h[1] = b;
  return u.v;
}

// 32-bit LDS byte offset of a __shared__ object (generic -> AS3 -> int)
static __device__ __forceinline__ uint32_t lds_off(const void* p) {
  return (uint32_t)(uintptr_t)(__attribute__((address_space(3))) const void*)p;
}

// Async DMA: 16 bytes global -> LDS per lane, tracked by ASYNCcnt, no VGPR data
static __device__ __forceinline__ void async_copy_b128(uint32_t dst_lds,
                                                       const void* src) {
  asm volatile("global_load_async_to_lds_b128 %0, %1, off"
               :: "v"(dst_lds), "v"((unsigned long long)(uintptr_t)src)
               : "memory");
}

static __device__ __forceinline__ void wait_async0() {
  asm volatile("s_wait_asynccnt 0x0" ::: "memory");
}

// --------------------------- rx[b] = rsqrt(sum x^2) ------------------------
__global__ __launch_bounds__(256) void norm_x_kernel(
    const float* __restrict__ x, float* __restrict__ rx)
{
  const int row  = blockIdx.x * 8 + (threadIdx.x >> 5);
  const int lane = threadIdx.x & 31;
  const float* p = x + (size_t)row * D_DIM;
  float s = 0.0f;
  #pragma unroll
  for (int i = lane; i < D_DIM; i += 32) { float v = p[i]; s += v * v; }
  #pragma unroll
  for (int off = 16; off > 0; off >>= 1) s += __shfl_xor(s, off, 32);
  if (lane == 0) rx[row] = rsqrtf(fmaxf(s, 1e-12f));
}

// --------------------------- rw[d] = rsqrt(sum W[d,:]^2) -------------------
__global__ __launch_bounds__(256) void norm_w_kernel(
    const float* __restrict__ W, float* __restrict__ rw)
{
  __shared__ float red[256];
  const int d = blockIdx.x;
  const float* p = W + (size_t)d * N_DIM;
  float s = 0.0f;
  for (int i = threadIdx.x; i < N_DIM; i += 256) { float v = p[i]; s += v * v; }
  red[threadIdx.x] = s;
  __syncthreads();
  #pragma unroll
  for (int off = 128; off > 0; off >>= 1) {
    if (threadIdx.x < off) red[threadIdx.x] += red[threadIdx.x + off];
    __syncthreads();
  }
  if (threadIdx.x == 0) rw[d] = rsqrtf(fmaxf(red[0], 1e-12f));
}

// --------------------- A' = x * rw  -> f16 hi/lo planes [M][K] -------------
__global__ __launch_bounds__(256) void conv_a_kernel(
    const float* __restrict__ x, const float* __restrict__ rw,
    _Float16* __restrict__ ahi, _Float16* __restrict__ alo)
{
  const int idx = blockIdx.x * 256 + threadIdx.x;
  const float v = x[idx] * rw[idx & (D_DIM - 1)];
  const _Float16 h = (_Float16)v;
  ahi[idx] = h;
  alo[idx] = (_Float16)((v - (float)h) * LO_SCALE);
}

// --------- Bt[n][k] = W[k][n]  (transposed, N padded) -> f16 hi/lo ---------
__global__ __launch_bounds__(256) void conv_b_kernel(
    const float* __restrict__ W,
    _Float16* __restrict__ bhi, _Float16* __restrict__ blo)
{
  __shared__ float t[32][33];
  const int n0 = blockIdx.x * 32;
  const int k0 = blockIdx.y * 32;
  const int tx = threadIdx.x;   // 0..31
  const int ty = threadIdx.y;   // 0..7
  #pragma unroll
  for (int i = 0; i < 32; i += 8) {
    const int k = k0 + ty + i, n = n0 + tx;
    t[ty + i][tx] = (n < N_DIM) ? W[(size_t)k * N_DIM + n] : 0.0f;
  }
  __syncthreads();
  #pragma unroll
  for (int i = 0; i < 32; i += 8) {
    const int n = n0 + ty + i, k = k0 + tx;
    const float v = t[tx][ty + i];
    const _Float16 h = (_Float16)v;
    bhi[(size_t)n * D_DIM + k] = h;
    blo[(size_t)n * D_DIM + k] = (_Float16)((v - (float)h) * LO_SCALE);
  }
}

// ------------------------------- main GEMM ---------------------------------
__global__ __launch_bounds__(256) void cosine_gemm_kernel(
    const _Float16* __restrict__ Ahi, const _Float16* __restrict__ Alo,
    const _Float16* __restrict__ Bhi, const _Float16* __restrict__ Blo,
    const float* __restrict__ rx, float* __restrict__ out)
{
  // double-buffered stages: [stage][plane hi/lo][row][padded K]
  __shared__ __align__(16) _Float16 lA[2][2][TM][LDSTR];
  __shared__ __align__(16) _Float16 lB[2][2][TN][LDSTR];

  const int bn    = blockIdx.x;
  const int bm    = blockIdx.y;
  const int tid   = threadIdx.x;
  const int lane  = tid & 31;
  const int wave  = tid >> 5;
  const int waveM = wave & 3;     // 4 waves along M (32 rows each)
  const int waveN = wave >> 2;    // 2 waves along N (32 cols each)
  const int g     = lane >> 4;    // half-wave group (ISA fragment layout)
  const int lm    = lane & 15;

  const int aRow0 = bm * TM;
  const int bRow0 = bn * TN;      // row index into transposed B (== n)

  v8f accH[2][2] = {};            // hi*hi
  v8f accL[2][2] = {};            // (hi*lo + lo*hi) * 2^12

  const int ar = tid >> 1, ah = tid & 1;   // A staging: 128 rows x 2 halves
  const int br = tid >> 2, bc = tid & 3;   // B staging: 64 rows x 4 chunks

  // per-thread async staging of one 32-wide K chunk into stage s
  auto stage = [&](int s, int k0) {
    const size_t ga = (size_t)(aRow0 + ar) * D_DIM + k0 + ah * 16;
    async_copy_b128(lds_off(&lA[s][0][ar][ah * 16]),     Ahi + ga);
    async_copy_b128(lds_off(&lA[s][0][ar][ah * 16 + 8]), Ahi + ga + 8);
    async_copy_b128(lds_off(&lA[s][1][ar][ah * 16]),     Alo + ga);
    async_copy_b128(lds_off(&lA[s][1][ar][ah * 16 + 8]), Alo + ga + 8);
    const size_t gb = (size_t)(bRow0 + br) * D_DIM + k0 + bc * 8;
    async_copy_b128(lds_off(&lB[s][0][br][bc * 8]), Bhi + gb);
    async_copy_b128(lds_off(&lB[s][1][br][bc * 8]), Blo + gb);
  };

  stage(0, 0);   // prologue fill of stage 0

  for (int i = 0; i < KSTEPS; ++i) {
    const int s = i & 1;

    wait_async0();      // my async fills of stage s (and any later) complete
    __syncthreads();    // everyone's fills of stage s visible; everyone done
                        // reading buffer s^1 (consumed by WMMAs last iter)

    if (i + 1 < KSTEPS) stage(s ^ 1, (i + 1) * 32);   // DMA next chunk now

    // ---- build fragments per ISA 16-bit layouts ----
    // A 16x32: group g elems 0..7 -> K=g*8.., elems 8..15 -> K=16+g*8..
    v16h aH[2], aL[2], bH[2], bL[2];
    #pragma unroll
    for (int mi = 0; mi < 2; ++mi) {
      const _Float16* b0 = &lA[s][0][waveM * 32 + mi * 16 + lm][0];
      const _Float16* b1 = &lA[s][1][waveM * 32 + mi * 16 + lm][0];
      aH[mi] = join8(*(const v8h*)(b0 + g * 8), *(const v8h*)(b0 + 16 + g * 8));
      aL[mi] = join8(*(const v8h*)(b1 + g * 8), *(const v8h*)(b1 + 16 + g * 8));
    }
    // B 32x16: group g elems 0..7 -> K=g*8.., elems 8..15 -> K=16+g*8.. for column n=lm
    #pragma unroll
    for (int ni = 0; ni < 2; ++ni) {
      const _Float16* b0 = &lB[s][0][waveN * 32 + ni * 16 + lm][0];
      const _Float16* b1 = &lB[s][1][waveN * 32 + ni * 16 + lm][0];
      bH[ni] = join8(*(const v8h*)(b0 + g * 8), *(const v8h*)(b0 + 16 + g * 8));
      bL[ni] = join8(*(const v8h*)(b1 + g * 8), *(const v8h*)(b1 + 16 + g * 8));
    }

    // ---- 3-product split-f16 accumulation (12 WMMAs / wave / K-step) ----
    #pragma unroll
    for (int mi = 0; mi < 2; ++mi)
      #pragma unroll
      for (int ni = 0; ni < 2; ++ni) {
        accH[mi][ni] = __builtin_amdgcn_wmma_f32_16x16x32_f16(
            false, aH[mi], false, bH[ni], (short)0, accH[mi][ni], false, false);
        accL[mi][ni] = __builtin_amdgcn_wmma_f32_16x16x32_f16(
            false, aH[mi], false, bL[ni], (short)0, accL[mi][ni], false, false);
        accL[mi][ni] = __builtin_amdgcn_wmma_f32_16x16x32_f16(
            false, aL[mi], false, bH[ni], (short)0, accL[mi][ni], false, false);
      }
  }

  // ---- epilogue: combine hi + lo*2^-12, scale by rx[row], bounds-store ----
  // C/D layout: lane group g, col n=lm, VGPR r -> row = g*8 + r
  #pragma unroll
  for (int mi = 0; mi < 2; ++mi) {
    const int rowBase = aRow0 + waveM * 32 + mi * 16 + g * 8;
    #pragma unroll
    for (int ni = 0; ni < 2; ++ni) {
      const int col = bRow0 + waveN * 32 + ni * 16 + lm;
      if (col < N_DIM) {
        #pragma unroll
        for (int r = 0; r < 8; ++r) {
          const int row = rowBase + r;
          out[(size_t)row * N_DIM + col] =
              (accH[mi][ni][r] + accL[mi][ni][r] * LO_SCALE_INV) * rx[row];
        }
      }
    }
  }
}

// ---------------------------------------------------------------------------
extern "C" void kernel_launch(void* const* d_in, const int* in_sizes, int n_in,
                              void* d_out, int out_size, void* d_ws, size_t ws_size,
                              hipStream_t stream) {
  (void)in_sizes; (void)n_in; (void)out_size; (void)ws_size;
  const float* x = (const float*)d_in[0];   // [4096, 512]
  const float* W = (const float*)d_in[1];   // [512, 5994]
  float* out = (float*)d_out;               // [4096, 5994]

  // workspace carve-up (~20.7 MB)
  float* rx = (float*)d_ws;                       // 4096 f32
  float* rw = rx + B_DIM;                         // 512 f32
  _Float16* ahi = (_Float16*)(rw + D_DIM);        // [M][K] f16
  _Float16* alo = ahi + (size_t)B_DIM * D_DIM;
  _Float16* bhi = alo + (size_t)B_DIM * D_DIM;    // [N_PAD][K] f16 (transposed)
  _Float16* blo = bhi + (size_t)N_PAD * D_DIM;

  norm_x_kernel<<<B_DIM / 8, 256, 0, stream>>>(x, rx);
  norm_w_kernel<<<D_DIM, 256, 0, stream>>>(W, rw);
  conv_a_kernel<<<(B_DIM * D_DIM) / 256, 256, 0, stream>>>(x, rw, ahi, alo);
  conv_b_kernel<<<dim3(N_PAD / 32, D_DIM / 32), dim3(32, 8), 0, stream>>>(W, bhi, blo);
  cosine_gemm_kernel<<<dim3(N_PAD / TN, B_DIM / TM), 256, 0, stream>>>(
      ahi, alo, bhi, blo, rx, out);
}
